// SparseAttentionExpert_5815385719071
// MI455X (gfx1250) — matrix operand order repaired and checked by reading, not compiled, not run on hardware
//
#include <hip/hip_runtime.h>

// ---------------------------------------------------------------------------
// Sparse (top-64) multi-head attention for MI455X (gfx1250, wave32, WMMA).
//   x[B,N,D] fp32, W* [D,D] fp32, b* [D] fp32  ->  out [B,N,D] fp32
// Pipeline: cast->bf16, QKV GEMMs (WMMA bf16; A tiles via async-LDS copies,
// B tiles via Tensor Data Mover DMA, double-buffered), fused
// scores/top-k/softmax/AV with LDS-resident score slabs, output projection.
// ---------------------------------------------------------------------------

#define DMODEL 1024
#define NHEAD  16
#define DHEAD  64
#define NSEQ   2048
#define BATCH  4
#define KATT   64
#define SCALE  0.125f      // 1/sqrt(64)
#define LDA    40          // padded LDS row stride (elems): 80B, breaks conflicts

typedef unsigned int uint32;
typedef __bf16 bf16;
typedef __attribute__((ext_vector_type(16))) __bf16 v16bf;
typedef __attribute__((ext_vector_type(8)))  __bf16 v8bf;
typedef __attribute__((ext_vector_type(8)))  float  v8f;

#ifndef __has_builtin
#define __has_builtin(x) 0
#endif
#if __has_builtin(__builtin_amdgcn_tensor_load_to_lds)
#define HAVE_TDM 1
#else
#define HAVE_TDM 0
#endif

// ---------------------------------------------------------------------------
// WMMA helper: D = A(16x32 bf16) * B(32x16 bf16) + C(16x16 f32)
// ---------------------------------------------------------------------------
__device__ __forceinline__ v8f wmma_bf16(v16bf a, v16bf b, v8f c) {
    return __builtin_amdgcn_wmma_f32_16x16x32_bf16(
        /*neg_a=*/false, a, /*neg_b=*/false, b,
        /*c_mod=*/(short)0, c, /*reuse_a=*/false, /*reuse_b=*/false);
}

// A-matrix 16x32 bf16 fragment (row stride ld elems).
// Lanes 0-15 row M=lane: K=0..7 | K=16..23; lanes 16-31: K=8..15 | K=24..31.
__device__ __forceinline__ v16bf load_fragA(const bf16* __restrict__ base, int ld) {
    const int lane = threadIdx.x & 31;
    const bf16* p = base + (size_t)(lane & 15) * ld + ((lane >> 4) << 3);
    v8bf lo = *(const v8bf*)(p);
    v8bf hi = *(const v8bf*)(p + 16);
    v16bf a;
#pragma unroll
    for (int i = 0; i < 8; ++i) { a[i] = lo[i]; a[i + 8] = hi[i]; }
    return a;
}

// B-matrix 32x16 bf16 fragment from row-major W[n][k] rows.
// Lanes 0-15 col N=lane: K=0..7 | K=16..23; lanes 16-31 col N=lane-16: K=8..15 | K=24..31.
__device__ __forceinline__ v16bf load_fragB(const bf16* __restrict__ base, int ld) {
    const int lane = threadIdx.x & 31;
    const bf16* p = base + (size_t)(lane & 15) * ld + ((lane >> 4) << 3);
    v8bf lo = *(const v8bf*)(p);
    v8bf hi = *(const v8bf*)(p + 16);
    v16bf b;
#pragma unroll
    for (int i = 0; i < 8; ++i) { b[i] = lo[i]; b[i + 8] = hi[i]; }
    return b;
}

// Async global->LDS 16-byte copy (ASYNCcnt-tracked, CDNA5).
// LDS offset = low 32 bits of the generic pointer (LDS aperture mapping).
__device__ __forceinline__ void async_copy_b128(const bf16* g, bf16* l) {
    unsigned ldsOff = (unsigned)(uintptr_t)l;
    unsigned long long ga = (unsigned long long)(uintptr_t)g;
    asm volatile("global_load_async_to_lds_b128 %0, %1, off"
                 :: "v"(ldsOff), "v"(ga) : "memory");
}
template <int N>
__device__ __forceinline__ void wait_async() {
    asm volatile("s_wait_asynccnt %0" :: "i"(N) : "memory");
}

#if HAVE_TDM
typedef __attribute__((ext_vector_type(4))) unsigned int u32x4;
typedef __attribute__((ext_vector_type(8))) int i32x8;
typedef __attribute__((ext_vector_type(4))) int i32x4;

// TDM DMA of one 64-row x 32-elem bf16 tile from a row-major [1024 x 1024]
// tensor into LDS, padding 16B after every 64B row (matches LDA=40 stride).
// TENSORcnt-tracked; descriptors must be wave-uniform (SGPR groups).
__device__ __forceinline__ void tdm_load_tile_b(const bf16* g, bf16* l) {
    unsigned long long ga = (unsigned long long)(uintptr_t)g;
    unsigned lo32 = (unsigned)(uintptr_t)l;
    u32x4 g0;
    g0[0] = 1u;                                                    // count=1, load
    g0[1] = (unsigned)__builtin_amdgcn_readfirstlane((int)lo32);   // lds_addr
    g0[2] = (unsigned)__builtin_amdgcn_readfirstlane((int)(unsigned)ga);
    g0[3] = (unsigned)__builtin_amdgcn_readfirstlane(
        (int)(((unsigned)(ga >> 32) & 0x01FFFFFFu) | 0x80000000u)); // addr hi | type=2
    i32x8 g1;
    g1[0] = (int)((1u << 16)      // data_size = 2B
                | (1u << 20)      // pad_enable
                | (3u << 22)      // pad_interval: every 16 DWORDs (64B row)
                | (3u << 25));    // pad_amount: 4 DWORDs (16B)
    g1[1] = (int)(1024u << 16);   // tensor_dim0 = 1024
    g1[2] = (int)(1024u << 16);   // tensor_dim1 = 1024
    g1[3] = (int)(32u << 16);     // tile_dim0 = 32 elems (64B)
    g1[4] = 64;                   // tile_dim1 = 64 rows
    g1[5] = 1024;                 // tensor_dim0_stride = 1024 elems
    g1[6] = 0;
    g1[7] = 0;
    __builtin_amdgcn_tensor_load_to_lds(g0, g1, (i32x4)0, (i32x4)0, (i32x8)0, 0);
}
#endif

// Monotonic float<->uint mapping for radix-style top-k on sortable keys.
__device__ __forceinline__ uint32 f2key(float f) {
    uint32 u = __float_as_uint(f);
    return u ^ (((uint32)((int)u >> 31)) | 0x80000000u);
}
__device__ __forceinline__ float key2f(uint32 u) {
    u = (u & 0x80000000u) ? (u ^ 0x80000000u) : ~u;
    return __uint_as_float(u);
}

// ---------------------------------------------------------------------------
// fp32 -> bf16 cast
// ---------------------------------------------------------------------------
__global__ void cast_f32_bf16(const float* __restrict__ in, bf16* __restrict__ out, int n) {
    int i = blockIdx.x * blockDim.x + threadIdx.x;
    int stride = gridDim.x * blockDim.x;
    for (; i < n; i += stride) out[i] = (bf16)in[i];
}

// ---------------------------------------------------------------------------
// GEMM: Y[M x 1024] = X[M x 1024] @ W^T (W row-major [1024 x 1024]) + bias.
// Block tile 128x64, 8 waves x (16 rows x 64 cols). K pipelined in 32-slabs:
// A staged by per-lane async-LDS copies, B staged by one TDM DMA per step,
// both double-buffered; WMMA fed from LDS.
// mode 0: scatter bf16 to [B,H,N,d];  mode 1: row-major fp32 to outP.
// ---------------------------------------------------------------------------
__global__ void __launch_bounds__(256)
gemm_xwT(const bf16* __restrict__ X, const bf16* __restrict__ W,
         const float* __restrict__ bias, bf16* __restrict__ outS,
         float* __restrict__ outP, int mode) {
    __shared__ bf16 Ab[2][128 * LDA];   // 2 x 10 KB
    __shared__ bf16 Bb[2][64 * LDA];    // 2 x  5 KB

    const int tid  = threadIdx.x;
    const int lane = tid & 31;
    const int wave = tid >> 5;
    const int rowBase0 = blockIdx.x * 128;
    const int rowBase  = rowBase0 + wave * 16;
    const int colBase  = blockIdx.y * 64;

    // A staging: 128 rows x 4 chunks of 16B -> two async ops per thread.
    const int sRow   = tid >> 2;
    const int sChunk = (tid & 3) << 3;          // element offset (8 bf16 = 16B)

    const bf16* Wtile = W + (size_t)colBase * DMODEL;   // B tile origin (k added)
    const bf16* Wc  = W + (size_t)(colBase + sRow) * DMODEL + sChunk;
    const bf16* Xa0 = X + (size_t)(rowBase0 + sRow)      * DMODEL + sChunk;
    const bf16* Xa1 = X + (size_t)(rowBase0 + sRow + 64) * DMODEL + sChunk;
    (void)Wc; (void)Wtile;

#if HAVE_TDM
#define ISSUE(buf, ks)                                                        \
    do {                                                                      \
        async_copy_b128(Xa0 + (ks), &Ab[buf][sRow * LDA + sChunk]);           \
        async_copy_b128(Xa1 + (ks), &Ab[buf][(sRow + 64) * LDA + sChunk]);    \
        if (wave == 0) tdm_load_tile_b(Wtile + (ks), &Bb[buf][0]);            \
    } while (0)
#define WAIT_FILLS()                                                          \
    do {                                                                      \
        wait_async<2>();                        /* prev A tile landed   */    \
        if (wave == 0) __builtin_amdgcn_s_wait_tensorcnt(1); /* prev B  */    \
    } while (0)
#else
#define ISSUE(buf, ks)                                                        \
    do {                                                                      \
        async_copy_b128(Wc  + (ks), &Bb[buf][sRow * LDA + sChunk]);           \
        async_copy_b128(Xa0 + (ks), &Ab[buf][sRow * LDA + sChunk]);           \
        async_copy_b128(Xa1 + (ks), &Ab[buf][(sRow + 64) * LDA + sChunk]);    \
    } while (0)
#define WAIT_FILLS() wait_async<3>()
#endif

    ISSUE(0, 0);

    v8f acc0 = {}, acc1 = {}, acc2 = {}, acc3 = {};
    for (int i = 0; i < 32; ++i) {
        const int cur = i & 1;
        ISSUE(cur ^ 1, ((i + 1) & 31) * 32);    // wrap keeps wait-count exact
        WAIT_FILLS();                            // previous buffer filled
        __syncthreads();

        v16bf a  = load_fragA(&Ab[cur][(wave * 16) * LDA], LDA);
        v16bf b0 = load_fragB(&Bb[cur][ 0 * LDA], LDA);
        v16bf b1 = load_fragB(&Bb[cur][16 * LDA], LDA);
        v16bf b2 = load_fragB(&Bb[cur][32 * LDA], LDA);
        v16bf b3 = load_fragB(&Bb[cur][48 * LDA], LDA);
        acc0 = wmma_bf16(a, b0, acc0);
        acc1 = wmma_bf16(a, b1, acc1);
        acc2 = wmma_bf16(a, b2, acc2);
        acc3 = wmma_bf16(a, b3, acc3);
        __syncthreads();                         // buf[cur] free for refill
    }
#undef ISSUE
#undef WAIT_FILLS

    const int rOff = (lane >> 4) << 3;
    const int nc = lane & 15;
    v8f accs[4] = {acc0, acc1, acc2, acc3};
#pragma unroll
    for (int t = 0; t < 4; ++t) {
        const int c = colBase + t * 16 + nc;
        const float bv = bias[c];
#pragma unroll
        for (int r = 0; r < 8; ++r) {
            const int m = rowBase + rOff + r;
            const float v = accs[t][r] + bv;
            if (mode == 0) {
                const int b = m / NSEQ, n = m - b * NSEQ;
                const int h = c >> 6, dd = c & 63;
                outS[(((size_t)b * NHEAD + h) * NSEQ + n) * DHEAD + dd] = (bf16)v;
            } else {
                outP[(size_t)m * DMODEL + c] = v;
            }
        }
    }
}

// ---------------------------------------------------------------------------
// Fused attention: per (b,h, 16-query tile):
//   1. scores = (Q Kt) * SCALE via WMMA, stored to LDS as sortable uint keys
//   2. exact top-64 per query via 32-step radix threshold search
//   3. softmax over selected + weighted V accumulation -> bf16 attn out
// ---------------------------------------------------------------------------
__global__ void __launch_bounds__(256)
attn_topk(const bf16* __restrict__ Q, const bf16* __restrict__ K,
          const bf16* __restrict__ V, bf16* __restrict__ AO) {
    extern __shared__ char smem[];
    uint32* skeys  = (uint32*)smem;                                   // 16*2048
    uint32* selIdx = (uint32*)(smem + 16 * NSEQ * 4);                 // 16*64
    float*  selW   = (float*)(smem + 16 * NSEQ * 4 + 16 * KATT * 4);  // 16*64
    uint32* cnt    = (uint32*)(smem + 16 * NSEQ * 4 + 16 * KATT * 8); // 16
    float*  rowMax = (float*)(smem + 16 * NSEQ * 4 + 16 * KATT * 8 + 64);

    const int tid  = threadIdx.x;
    const int lane = tid & 31;
    const int wave = tid >> 5;
    const int bh   = blockIdx.y;
    const int qbase = blockIdx.x * 16;

    if (tid < 16) cnt[tid] = 0;

    // --- phase 1: WMMA scores -> LDS sortable keys --------------------------
    const bf16* Qt = Q + ((size_t)bh * NSEQ + qbase) * DHEAD;
    v16bf qa0 = load_fragA(Qt, DHEAD);        // k = 0..31
    v16bf qa1 = load_fragA(Qt + 32, DHEAD);   // k = 32..63

    for (int kt = 0; kt < 16; ++kt) {
        const int keybase = kt * 128 + wave * 16;
        const bf16* Kt = K + ((size_t)bh * NSEQ + keybase) * DHEAD;
        v16bf kb0 = load_fragB(Kt, DHEAD);
        v16bf kb1 = load_fragB(Kt + 32, DHEAD);
        v8f acc = {};
        acc = wmma_bf16(qa0, kb0, acc);
        acc = wmma_bf16(qa1, kb1, acc);
        const int row0 = (lane >> 4) << 3;
        const int col = keybase + (lane & 15);
#pragma unroll
        for (int r = 0; r < 8; ++r)
            skeys[(row0 + r) * NSEQ + col] = f2key(acc[r] * SCALE);
    }
    __syncthreads();

    // --- phase 2: exact top-64 select per query -----------------------------
    for (int qi = 0; qi < 2; ++qi) {
        const int q = wave + qi * 8;
        const uint32* kq = skeys + (size_t)q * NSEQ;
        uint32 kr[64];
        uint32 mx = 0;
#pragma unroll
        for (int i = 0; i < 64; ++i) {
            kr[i] = kq[i * 32 + lane];        // bank-conflict-free stride 32
            mx = kr[i] > mx ? kr[i] : mx;
        }
#pragma unroll
        for (int off = 16; off > 0; off >>= 1) {
            uint32 o = (uint32)__shfl_xor((int)mx, off, 32);
            mx = o > mx ? o : mx;
        }
        // largest T with count(keys >= T) >= 64  (== 64th largest key)
        uint32 T = 0;
        for (int bit = 31; bit >= 0; --bit) {
            const uint32 trial = T | (1u << bit);
            int c = 0;
#pragma unroll
            for (int i = 0; i < 64; ++i) c += (kr[i] >= trial) ? 1 : 0;
#pragma unroll
            for (int off = 16; off > 0; off >>= 1) c += __shfl_xor(c, off, 32);
            if (c >= KATT) T = trial;
        }
        if (lane == 0) rowMax[q] = key2f(mx);
        // gather selected (value,index) pairs, ties capped at 64
#pragma unroll
        for (int i = 0; i < 64; ++i) {
            const uint32 kv = kr[i];
            if (kv >= T) {
                const uint32 pos = atomicAdd(&cnt[q], 1u);
                if (pos < (uint32)KATT) {
                    selIdx[q * KATT + pos] = (uint32)(i * 32 + lane);
                    selW[q * KATT + pos]   = key2f(kv);
                }
            }
        }
    }
    __syncthreads();

    // --- phase 3: softmax over 64 + weighted V gather -----------------------
    const int b = bh / NHEAD, h = bh - b * NHEAD;
    for (int qi = 0; qi < 2; ++qi) {
        const int q = wave + qi * 8;
        const float mF = rowMax[q];
        const float w0 = __expf(selW[q * KATT + lane] - mF);
        const float w1 = __expf(selW[q * KATT + lane + 32] - mF);
        float s = w0 + w1;
#pragma unroll
        for (int off = 16; off > 0; off >>= 1) s += __shfl_xor(s, off, 32);
        const float inv = 1.0f / s;
        selW[q * KATT + lane] = w0;
        selW[q * KATT + lane + 32] = w1;
        __syncthreads();   // uniform: every wave runs both iterations

        float a0 = 0.f, a1 = 0.f;
#pragma unroll 8
        for (int j = 0; j < 64; ++j) {
            const float wj = selW[q * KATT + j];
            const uint32 idx = selIdx[q * KATT + j];
            const bf16* vr = V + ((size_t)bh * NSEQ + idx) * DHEAD;
            a0 += wj * (float)vr[lane];
            a1 += wj * (float)vr[lane + 32];
        }
        const int n = qbase + q;
        bf16* o = AO + ((size_t)b * NSEQ + n) * DMODEL + h * DHEAD;
        o[lane]      = (bf16)(a0 * inv);
        o[lane + 32] = (bf16)(a1 * inv);
    }
}

// ---------------------------------------------------------------------------
// Host launcher
// ---------------------------------------------------------------------------
extern "C" void kernel_launch(void* const* d_in, const int* in_sizes, int n_in,
                              void* d_out, int out_size, void* d_ws, size_t ws_size,
                              hipStream_t stream) {
    (void)in_sizes; (void)n_in; (void)out_size; (void)ws_size;

    const float* x  = (const float*)d_in[0];
    const float* Wq = (const float*)d_in[1];
    const float* bq = (const float*)d_in[2];
    const float* Wk = (const float*)d_in[3];
    const float* bk = (const float*)d_in[4];
    const float* Wv = (const float*)d_in[5];
    const float* bv = (const float*)d_in[6];
    const float* Wo = (const float*)d_in[7];
    const float* bo = (const float*)d_in[8];

    const size_t M   = (size_t)BATCH * NSEQ;        // 8192 tokens
    const size_t SZx = M * DMODEL * 2;              // bf16 activations (16 MB)
    const size_t SZw = (size_t)DMODEL * DMODEL * 2; // bf16 weights (2 MB)

    char* ws = (char*)d_ws;
    bf16* xb  = (bf16*)ws;            ws += SZx;
    bf16* Wqb = (bf16*)ws;            ws += SZw;
    bf16* Wkb = (bf16*)ws;            ws += SZw;
    bf16* Wvb = (bf16*)ws;            ws += SZw;
    bf16* Wob = (bf16*)ws;            ws += SZw;
    bf16* Qb  = (bf16*)ws;            ws += SZx;    // [B,H,N,d]
    bf16* Kb  = (bf16*)ws;            ws += SZx;    // [B,H,N,d]
    bf16* Vb  = (bf16*)ws;            ws += SZx;    // [B,H,N,d]
    bf16* AO  = (bf16*)ws;            ws += SZx;    // [B,N,D]

    const int nX = (int)(M * DMODEL);
    const int nW = DMODEL * DMODEL;
    cast_f32_bf16<<<4096, 256, 0, stream>>>(x,  xb,  nX);
    cast_f32_bf16<<<2048, 256, 0, stream>>>(Wq, Wqb, nW);
    cast_f32_bf16<<<2048, 256, 0, stream>>>(Wk, Wkb, nW);
    cast_f32_bf16<<<2048, 256, 0, stream>>>(Wv, Wvb, nW);
    cast_f32_bf16<<<2048, 256, 0, stream>>>(Wo, Wob, nW);

    dim3 gg((unsigned)(M / 128), DMODEL / 64);
    gemm_xwT<<<gg, 256, 0, stream>>>(xb, Wqb, bq, Qb, nullptr, 0);
    gemm_xwT<<<gg, 256, 0, stream>>>(xb, Wkb, bk, Kb, nullptr, 0);
    gemm_xwT<<<gg, 256, 0, stream>>>(xb, Wvb, bv, Vb, nullptr, 0);

    const size_t smem = (size_t)16 * NSEQ * 4      // score keys (128 KB)
                      + (size_t)16 * KATT * 4      // selected indices
                      + (size_t)16 * KATT * 4      // selected weights
                      + 64 + 64;                   // counters + row max
    attn_topk<<<dim3(NSEQ / 16, BATCH * NHEAD), 256, smem, stream>>>(Qb, Kb, Vb, AO);

    gemm_xwT<<<gg, 256, 0, stream>>>(AO, Wob, bo, nullptr, (float*)d_out, 1);
}
